// MSDeformAttn_19791209300006
// MI455X (gfx1250) — compile-verified, matrix operand order checked by reading, not run on hardware
//
#include <hip/hip_runtime.h>
#include <hip/hip_bf16.h>

// MSDeformAttn for MI455X (gfx1250, wave32).
// GEMMs via v_wmma_f32_16x16x32_f16 (f16 in, fp32 accumulate) with the weight
// slab staged in LDS (ds_load fragments) and double-buffered A fragments.
// Softmax + bilinear deformable sampling in fp32 VALU.

typedef __attribute__((ext_vector_type(16))) _Float16 v16h;
typedef __attribute__((ext_vector_type(8)))  float    v8f;

#define D_MODEL 256
#define NHEADS  8
#define NLEV    4
#define NPTS    4
#define HDIM    32   // D_MODEL / NHEADS
#define KMAX    256  // K of every GEMM in this model

// ---------------------------------------------------------------- converts
__global__ __launch_bounds__(256) void k_cvt_half(const float* __restrict__ src,
                                                  _Float16* __restrict__ dst,
                                                  long n) {
  long i = (long)blockIdx.x * blockDim.x + threadIdx.x;
  long stride = (long)gridDim.x * blockDim.x;
  for (; i < n; i += stride) dst[i] = (_Float16)src[i];
}

// src is [K][N] row-major fp32, dst is [N][K] row-major f16 (transposed).
__global__ __launch_bounds__(256) void k_cvt_tr_half(const float* __restrict__ src,
                                                     _Float16* __restrict__ dst,
                                                     int K, int N) {
  long n = (long)K * N;
  long i = (long)blockIdx.x * blockDim.x + threadIdx.x;
  long stride = (long)gridDim.x * blockDim.x;
  for (; i < n; i += stride) {
    int k = (int)(i / N);
    int c = (int)(i % N);
    dst[(long)c * K + k] = (_Float16)src[i];
  }
}

// ---------------------------------------------------------------- WMMA GEMM
// C[M,N] = A[M,K](f16) * B[K,N] + bias, with B given transposed as Bt[N][K] f16.
// K == 256, N % 64 == 0. Block = 4 waves; block computes a 64-row x 64-col
// super-tile: wave w -> rows [.. +16), all waves share one LDS weight slab.
__global__ __launch_bounds__(128) void k_gemm_wmma(const _Float16* __restrict__ A,
                                                   const _Float16* __restrict__ Bt,
                                                   const float* __restrict__ bias,
                                                   float* __restrict__ C,
                                                   int M, int N, int K) {
  __shared__ _Float16 bs[64 * KMAX];        // 32 KB weight slab

  const int lane = threadIdx.x & 31;
  const int wave = threadIdx.x >> 5;
  const int tm   = blockIdx.x * 4 + wave;   // 16-row tile index
  const int cn0  = blockIdx.y * 64;         // first column of block's slab

  // ---- stage Bt[cn0 .. cn0+64) x K into LDS (contiguous 64*K region) ----
  {
    const uint4* src = (const uint4*)(Bt + (size_t)cn0 * K);
    uint4* dst = (uint4*)bs;
    const int total = 64 * K / 8;           // uint4 elements
    for (int i = threadIdx.x; i < total; i += 128) dst[i] = src[i];
  }
  __syncthreads();

  const int l15     = lane & 15;
  const int halfsel = (lane < 16) ? 0 : 8;  // 16-bit A/B K striping per ISA

  int row  = tm * 16 + l15;
  int arow = row < M ? row : (M - 1);       // clamp: A-row m only feeds C-row m
  const _Float16* ap = A + (size_t)arow * K + halfsel;

  v8f acc[4] = {};

  // Double-buffered A fragment: issue next chunk's loads before current WMMAs.
  uint4 a0 = *(const uint4*)(ap);
  uint4 a1 = *(const uint4*)(ap + 16);

#pragma unroll
  for (int kt = 0; kt < KMAX; kt += 32) {
    v16h a;
    ((uint4*)&a)[0] = a0;
    ((uint4*)&a)[1] = a1;
    if (kt + 32 < KMAX) {
      a0 = *(const uint4*)(ap + kt + 32);
      a1 = *(const uint4*)(ap + kt + 48);
    }
    const int kb = kt + halfsel;
#pragma unroll
    for (int j = 0; j < 4; ++j) {
      v16h b;
      const _Float16* bp = bs + (j * 16 + l15) * KMAX + kb;   // LDS
      ((uint4*)&b)[0] = *(const uint4*)(bp);
      ((uint4*)&b)[1] = *(const uint4*)(bp + 16);
      acc[j] = __builtin_amdgcn_wmma_f32_16x16x32_f16(
          /*neg_a=*/false, a, /*neg_b=*/false, b,
          /*c_mod=*/(short)0, acc[j], /*reuse_a=*/false, /*reuse_b=*/false);
    }
  }

  // C/D layout: VGPR r -> row r + (lane<16 ? 0 : 8); column = lane%16
  const int rbase = tm * 16 + ((lane < 16) ? 0 : 8);
#pragma unroll
  for (int j = 0; j < 4; ++j) {
    const int col = cn0 + j * 16 + l15;
    const float bs_ = bias ? bias[col] : 0.0f;
#pragma unroll
    for (int r = 0; r < 8; ++r) {
      const int rr = rbase + r;
      if (rr < M) C[(size_t)rr * N + col] = acc[j][r] + bs_;
    }
  }
}

// ---------------------------------------------------------------- softmax(16)
// In-place softmax over contiguous groups of 16 (the L*P axis).
__global__ __launch_bounds__(256) void k_softmax16(float* __restrict__ a, int total) {
  int i = blockIdx.x * blockDim.x + threadIdx.x;
  if (i >= total) return;
  float* p = a + (size_t)i * 16;
  float m = p[0];
#pragma unroll
  for (int j = 1; j < 16; ++j) m = fmaxf(m, p[j]);
  float e[16];
  float s = 0.0f;
#pragma unroll
  for (int j = 0; j < 16; ++j) { e[j] = __expf(p[j] - m); s += e[j]; }
  const float inv = 1.0f / s;
#pragma unroll
  for (int j = 0; j < 16; ++j) p[j] = e[j] * inv;
}

// ---------------------------------------------------------------- sampling
// One block per query; thread = (head h, channel c). value is [Len_in][256]
// head-major in channels. off is [Lq][H*L*P*2], attn is [Lq][H*L*P] (softmaxed).
__global__ __launch_bounds__(256) void k_sample(const float* __restrict__ value,
                                                const float* __restrict__ refp,
                                                const float* __restrict__ offv,
                                                const float* __restrict__ attn,
                                                const int* __restrict__ shapes,
                                                const int* __restrict__ lstart,
                                                float* __restrict__ out) {
  const int q = blockIdx.x;
  const int h = threadIdx.x >> 5;
  const int c = threadIdx.x & 31;

  const float* offq = offv + (size_t)q * (NHEADS * NLEV * NPTS * 2);
  const float* attq = attn + (size_t)q * (NHEADS * NLEV * NPTS);
  const float* refq = refp + (size_t)q * (NLEV * 2);

  float acc = 0.0f;
#pragma unroll
  for (int l = 0; l < NLEV; ++l) {
    const int Hs = shapes[l * 2 + 0];
    const int Ws = shapes[l * 2 + 1];
    const float rx = refq[l * 2 + 0];
    const float ry = refq[l * 2 + 1];
    const float* vbase = value + (size_t)lstart[l] * D_MODEL + h * HDIM + c;
#pragma unroll
    for (int p = 0; p < NPTS; ++p) {
      const int fi = (h * NLEV + l) * NPTS + p;
      // loc = ref + off/norm; x = loc.x*W - 0.5 = ref.x*W + off.x - 0.5
      const float x = rx * (float)Ws + offq[fi * 2 + 0] - 0.5f;
      const float y = ry * (float)Hs + offq[fi * 2 + 1] - 0.5f;
      const float aw = attq[fi];
      const float x0 = floorf(x), y0 = floorf(y);
      const float lw = x - x0, lh = y - y0;
      const int xi = (int)x0, yi = (int)y0;
#pragma unroll
      for (int dy = 0; dy < 2; ++dy) {
#pragma unroll
        for (int dx = 0; dx < 2; ++dx) {
          const int hc = yi + dy, wc = xi + dx;
          if (hc >= 0 && hc < Hs && wc >= 0 && wc < Ws) {
            const float wgt = (dy ? lh : 1.0f - lh) * (dx ? lw : 1.0f - lw);
            acc += aw * wgt * vbase[((size_t)hc * Ws + wc) * D_MODEL];
          }
        }
      }
    }
  }
  out[(size_t)q * D_MODEL + threadIdx.x] = acc;
}

// ---------------------------------------------------------------- launch
extern "C" void kernel_launch(void* const* d_in, const int* in_sizes, int n_in,
                              void* d_out, int out_size, void* d_ws, size_t ws_size,
                              hipStream_t stream) {
  (void)n_in; (void)out_size; (void)ws_size;
  const float* query  = (const float*)d_in[0];
  const float* refp   = (const float*)d_in[1];
  const float* inp    = (const float*)d_in[2];
  const int*   shapes = (const int*)d_in[3];
  const int*   lstart = (const int*)d_in[4];
  const float* W_off  = (const float*)d_in[5];
  const float* b_off  = (const float*)d_in[6];
  const float* W_attn = (const float*)d_in[7];
  const float* b_attn = (const float*)d_in[8];
  const float* W_v    = (const float*)d_in[9];
  const float* b_v    = (const float*)d_in[10];
  const float* W_out  = (const float*)d_in[11];
  const float* b_out  = (const float*)d_in[12];
  float* outp = (float*)d_out;

  const int Lq  = in_sizes[0] / D_MODEL;   // 8500
  const int Lin = in_sizes[2] / D_MODEL;   // 8500

  // -------- workspace carve-up (deterministic, 256B aligned) --------
  char* ws = (char*)d_ws;
  size_t o = 0;
  auto take = [&](size_t bytes) -> char* {
    char* p = ws + o;
    o += (bytes + 255) & ~(size_t)255;
    return p;
  };
  float*    f_value = (float*)take((size_t)Lin * D_MODEL * sizeof(float));
  float*    f_off   = (float*)take((size_t)Lq * (NHEADS * NLEV * NPTS * 2) * sizeof(float));
  float*    f_attn  = (float*)take((size_t)Lq * (NHEADS * NLEV * NPTS) * sizeof(float));
  float*    f_samp  = (float*)take((size_t)Lq * D_MODEL * sizeof(float));
  _Float16* h_inp   = (_Float16*)take((size_t)Lin * D_MODEL * sizeof(_Float16));
  _Float16* h_qry   = (_Float16*)take((size_t)Lq * D_MODEL * sizeof(_Float16));
  _Float16* h_samp  = (_Float16*)take((size_t)Lq * D_MODEL * sizeof(_Float16));
  _Float16* h_Wv    = (_Float16*)take((size_t)D_MODEL * D_MODEL * sizeof(_Float16));
  _Float16* h_Woff  = (_Float16*)take((size_t)D_MODEL * D_MODEL * sizeof(_Float16));
  _Float16* h_Wattn = (_Float16*)take((size_t)D_MODEL * 128 * sizeof(_Float16));
  _Float16* h_Wout  = (_Float16*)take((size_t)D_MODEL * D_MODEL * sizeof(_Float16));

  // -------- 1) fp32 -> f16 conversions --------
  k_cvt_half<<<2048, 256, 0, stream>>>(inp,   h_inp, (long)Lin * D_MODEL);
  k_cvt_half<<<2048, 256, 0, stream>>>(query, h_qry, (long)Lq * D_MODEL);
  k_cvt_tr_half<<<256, 256, 0, stream>>>(W_v,    h_Wv,    D_MODEL, D_MODEL);
  k_cvt_tr_half<<<256, 256, 0, stream>>>(W_off,  h_Woff,  D_MODEL, D_MODEL);
  k_cvt_tr_half<<<128, 256, 0, stream>>>(W_attn, h_Wattn, D_MODEL, 128);
  k_cvt_tr_half<<<256, 256, 0, stream>>>(W_out,  h_Wout,  D_MODEL, D_MODEL);

  // -------- 2) WMMA GEMMs (grid.x covers 4 M-tiles per block) --------
  const int mtiles  = (Lq + 15) / 16;      // == Lin tiles too
  const int gx      = (mtiles + 3) / 4;
  dim3 blk(128);

  // value = X @ W_v + b_v
  k_gemm_wmma<<<dim3(gx, D_MODEL / 64), blk, 0, stream>>>(
      h_inp, h_Wv, b_v, f_value, Lin, D_MODEL, D_MODEL);
  // off = Q @ W_off + b_off
  k_gemm_wmma<<<dim3(gx, D_MODEL / 64), blk, 0, stream>>>(
      h_qry, h_Woff, b_off, f_off, Lq, D_MODEL, D_MODEL);
  // attn logits = Q @ W_attn + b_attn   (N = 128)
  k_gemm_wmma<<<dim3(gx, 128 / 64), blk, 0, stream>>>(
      h_qry, h_Wattn, b_attn, f_attn, Lq, 128, D_MODEL);

  // -------- 3) softmax over L*P = 16 --------
  const int srows = Lq * NHEADS;
  k_softmax16<<<(srows + 255) / 256, 256, 0, stream>>>(f_attn, srows);

  // -------- 4) deformable bilinear sampling --------
  k_sample<<<Lq, 256, 0, stream>>>(f_value, refp, f_off, f_attn,
                                   shapes, lstart, f_samp);

  // -------- 5) output projection --------
  k_cvt_half<<<2048, 256, 0, stream>>>(f_samp, h_samp, (long)Lq * D_MODEL);
  k_gemm_wmma<<<dim3(gx, D_MODEL / 64), blk, 0, stream>>>(
      h_samp, h_Wout, b_out, outp, Lq, D_MODEL, D_MODEL);
}
